// SelfAttention_27393301414214
// MI455X (gfx1250) — matrix operand order repaired and checked by reading, not compiled, not run on hardware
//
#include <hip/hip_runtime.h>
#include <hip/hip_bf16.h>

// CDNA5 / gfx1250, wave32. Matrix ops via v_wmma_f32_16x16x32_bf16 with
// split-bf16 (hi/lo) operands for ~f32 accuracy at bf16 WMMA throughput.
// Attention staging uses the gfx1250 async global->LDS DMA path (ASYNCcnt).

typedef __attribute__((ext_vector_type(16))) __bf16 v16bf;
typedef __attribute__((ext_vector_type(8)))  float  v8f;
typedef __attribute__((ext_vector_type(2)))  __bf16 bf16x2;
typedef __attribute__((ext_vector_type(4)))  __bf16 bf16x4;

static __device__ __forceinline__ void split2(float x, __bf16& hi, __bf16& lo) {
  hi = (__bf16)x;                 // RNE truncation to bf16
  lo = (__bf16)(x - (float)hi);   // residual, |lo| <= 2^-9 |x|
}

// Wave-relative LDS byte offset of a generic __shared__ pointer:
// generic LDS addresses are {SHARED_BASE[63:32], lds_offset[31:0]} and the
// async-LDS VDST operand wants exactly addr[31:0] (HW adds LDS_BASE).
static __device__ __forceinline__ unsigned lds_off(const void* p) {
  return (unsigned)(unsigned long long)p;
}

static __device__ __forceinline__ void async_g2l_b128(unsigned lds_byte_off,
                                                      const void* gaddr) {
  asm volatile("global_load_async_to_lds_b128 %0, %1, off"
               :: "v"(lds_byte_off), "v"(gaddr) : "memory");
}

static __device__ __forceinline__ void wait_async0() {
  asm volatile("s_wait_asynccnt 0x0" ::: "memory");
}

// ---------------------------------------------------------------------------
// GEMM: C[M][Nn] = A[M][K] * Bt[Nn][K]^T + bias[Nn]   (both ops K-contiguous)
// Workgroup: 256 threads = 8 waves; 128x128 output tile; K step 32.
// Wave (mw = w&3, nw = w>>2) owns 32(M) x 64(N): 2 x 4 WMMA frags.
// LDS tiles pre-split into bf16 hi/lo planes (pitch 36: conflict-free,
// 8B-aligned b64 stores, 4B-aligned pair gathers).
// ---------------------------------------------------------------------------
#define GP 36

__global__ __launch_bounds__(256)
void gemm_bt_bias(const float* __restrict__ A, const float* __restrict__ Bt,
                  const float* __restrict__ bias, float* __restrict__ C,
                  int M, int Nn, int K)
{
  __shared__ __bf16 Ahs[128 * GP];
  __shared__ __bf16 Als[128 * GP];
  __shared__ __bf16 Bhs[128 * GP];
  __shared__ __bf16 Bls[128 * GP];

  const int tid  = threadIdx.x;
  const int lane = tid & 31;
  const int wave = tid >> 5;
  const int mw   = wave & 3;   // 4 M-bands of 32 rows
  const int nw   = wave >> 2;  // 2 N-bands of 64 cols
  const int half = lane >> 4;  // 0: lanes 0-15, 1: lanes 16-31
  const int lr   = lane & 15;
  const long mbase = (long)blockIdx.x * 128;
  const long nbase = (long)blockIdx.y * 128;

  v8f acc[2][4];
  #pragma unroll
  for (int mi = 0; mi < 2; ++mi)
    #pragma unroll
    for (int ni = 0; ni < 4; ++ni)
      #pragma unroll
      for (int e = 0; e < 8; ++e) acc[mi][ni][e] = 0.0f;

  const int r0 = tid >> 3;        // 0..31
  const int c4 = (tid & 7) * 4;   // 0,4,...,28

  for (int kb = 0; kb < K; kb += 32) {
    // Prefetch next K-step lines into cache while this tile computes.
    if (kb + 32 < K) {
      __builtin_prefetch(A  + (mbase + r0) * (long)K + (kb + 32) + c4, 0, 3);
      __builtin_prefetch(Bt + (nbase + r0) * (long)K + (kb + 32) + c4, 0, 3);
    }
    __syncthreads();
    // Cooperative load + split: each thread 4 float4 rows per tile.
    #pragma unroll
    for (int i = 0; i < 4; ++i) {
      int r = r0 + 32 * i;
      float4 a = *(const float4*)(A  + (mbase + r) * (long)K + kb + c4);
      float4 b = *(const float4*)(Bt + (nbase + r) * (long)K + kb + c4);
      bf16x4 h, l;
      __bf16 th, tl;
      split2(a.x, th, tl); h[0] = th; l[0] = tl;
      split2(a.y, th, tl); h[1] = th; l[1] = tl;
      split2(a.z, th, tl); h[2] = th; l[2] = tl;
      split2(a.w, th, tl); h[3] = th; l[3] = tl;
      *(bf16x4*)(Ahs + r * GP + c4) = h;
      *(bf16x4*)(Als + r * GP + c4) = l;
      split2(b.x, th, tl); h[0] = th; l[0] = tl;
      split2(b.y, th, tl); h[1] = th; l[1] = tl;
      split2(b.z, th, tl); h[2] = th; l[2] = tl;
      split2(b.w, th, tl); h[3] = th; l[3] = tl;
      *(bf16x4*)(Bhs + r * GP + c4) = h;
      *(bf16x4*)(Bls + r * GP + c4) = l;
    }
    __syncthreads();

    // A fragments: 16-bit A 16x32 ISA layout.
    // lane L: M = L%16; vgpr v pair K = ((v>=4)?16:0) + (L>=16?8:0) + 2*(v&3)
    v16bf afh[2], afl[2];
    #pragma unroll
    for (int mi = 0; mi < 2; ++mi) {
      int Mr = mw * 32 + mi * 16 + lr;
      #pragma unroll
      for (int v = 0; v < 8; ++v) {
        int k0 = ((v >= 4) ? 16 : 0) + half * 8 + 2 * (v & 3);
        bf16x2 th = *(const bf16x2*)(Ahs + Mr * GP + k0);
        bf16x2 tl = *(const bf16x2*)(Als + Mr * GP + k0);
        afh[mi][2*v] = th[0]; afh[mi][2*v+1] = th[1];
        afl[mi][2*v] = tl[0]; afl[mi][2*v+1] = tl[1];
      }
    }
    // B fragments: 16-bit B 32x16 layout.
    // lane L: N = L%16; vgpr v pair K = ((v>=4)?16:0) + (L>=16?8:0) + 2*(v&3)
    #pragma unroll
    for (int ni = 0; ni < 4; ++ni) {
      int Nr = nw * 64 + ni * 16 + lr;
      v16bf bfh, bfl;
      #pragma unroll
      for (int v = 0; v < 8; ++v) {
        int k0 = ((v >= 4) ? 16 : 0) + half * 8 + 2 * (v & 3);
        bf16x2 th = *(const bf16x2*)(Bhs + Nr * GP + k0);
        bf16x2 tl = *(const bf16x2*)(Bls + Nr * GP + k0);
        bfh[2*v] = th[0]; bfh[2*v+1] = th[1];
        bfl[2*v] = tl[0]; bfl[2*v+1] = tl[1];
      }
      #pragma unroll
      for (int mi = 0; mi < 2; ++mi) {
        acc[mi][ni] = __builtin_amdgcn_wmma_f32_16x16x32_bf16(
            false, afh[mi], false, bfh, (short)0, acc[mi][ni], false, false);
        acc[mi][ni] = __builtin_amdgcn_wmma_f32_16x16x32_bf16(
            false, afh[mi], false, bfl, (short)0, acc[mi][ni], false, false);
        acc[mi][ni] = __builtin_amdgcn_wmma_f32_16x16x32_bf16(
            false, afl[mi], false, bfh, (short)0, acc[mi][ni], false, false);
      }
    }
  }

  // Epilogue: C/D layout — vgpr r: M = r + (lane>=16?8:0), N = lane%16.
  #pragma unroll
  for (int ni = 0; ni < 4; ++ni) {
    long ncol = nbase + nw * 64 + ni * 16 + lr;
    float bv = bias[ncol];
    #pragma unroll
    for (int mi = 0; mi < 2; ++mi) {
      long mb = mbase + mw * 32 + mi * 16 + half * 8;
      #pragma unroll
      for (int r = 0; r < 8; ++r)
        C[(mb + r) * (long)Nn + ncol] = acc[mi][ni][r] + bv;
    }
  }
}

// ---------------------------------------------------------------------------
// Attention: one wave (one 32-thread workgroup) per (n, s') position.
//   S[h][e] = sum_d cq[h][d]*k[e][d] / 8 ; mask==0 -> -1e20 ; softmax over e
//   O[h][d] = sum_e P[h][e]*v[e][d]
// cq = queries row (s' even) or cond_feat row (s' odd).
// Staging: async global->LDS DMA (GLOBAL_LOAD_ASYNC_TO_LDS_B128, ASYNCcnt).
// ---------------------------------------------------------------------------
#define AP 68   // LDS pitch for 16x64 f32 rows (conflict-free, 16B aligned)
#define PP 20   // LDS pitch for 16x16 P tile

__global__ __launch_bounds__(32)
void attn16(const float* __restrict__ values, const float* __restrict__ keys,
            const float* __restrict__ queries, const int* __restrict__ mask,
            const float* __restrict__ cond_feat, float* __restrict__ attout)
{
  __shared__ float sCQ[16 * AP];
  __shared__ float sK [16 * AP];
  __shared__ float sV [16 * AP];
  __shared__ float sP [16 * PP];

  const int lane = threadIdx.x;
  const int half = lane >> 4;
  const int lr   = lane & 15;
  const int p    = blockIdx.x;        // 0 .. N*4096-1
  const int n    = p >> 12;
  const int s2   = p & 4095;

  const float* cqsrc = (s2 & 1)
      ? (cond_feat + ((long)n * 2048 + (s2 >> 1)) * 1024)
      : (queries   + ((long)n * 2048 + (s2 >> 1)) * 1024);
  const float* ksrc = keys   + ((long)n * 4096 + s2) * 1024;
  const float* vsrc = values + ((long)n * 4096 + s2) * 1024;

  // Stage 16x64 row-blocks into LDS via async LDS-DMA (24 b128 ops/lane-set).
  #pragma unroll
  for (int i = 0; i < 8; ++i) {
    int q   = lane + i * 32;     // float4 index 0..255
    int row = q >> 4;
    int cc  = (q & 15) * 4;
    async_g2l_b128(lds_off(sCQ + row * AP + cc), cqsrc + q * 4);
    async_g2l_b128(lds_off(sK  + row * AP + cc), ksrc  + q * 4);
    async_g2l_b128(lds_off(sV  + row * AP + cc), vsrc  + q * 4);
  }
  wait_async0();     // ASYNCcnt == 0: LDS writes landed
  __syncthreads();   // single-wave WG: cheap; orders DS

  // ---- S = cq * k^T (M=h, N=e, K=d=64), two K=32 steps, split-bf16 x3
  v8f S;
  #pragma unroll
  for (int e = 0; e < 8; ++e) S[e] = 0.0f;

  #pragma unroll
  for (int ks = 0; ks < 2; ++ks) {
    v16bf ah, al, bh, bl;
    #pragma unroll
    for (int v = 0; v < 8; ++v) {
      int ka = ks * 32 + ((v >= 4) ? 16 : 0) + half * 8 + 2 * (v & 3);
      __bf16 th, tl;
      split2(sCQ[lr * AP + ka],     th, tl); ah[2*v]   = th; al[2*v]   = tl;
      split2(sCQ[lr * AP + ka + 1], th, tl); ah[2*v+1] = th; al[2*v+1] = tl;
      int kbk = ks * 32 + ((v >= 4) ? 16 : 0) + half * 8 + 2 * (v & 3);     // B[K=d][N=e] = k[e][d]
      split2(sK[lr * AP + kbk],     th, tl); bh[2*v]   = th; bl[2*v]   = tl;
      split2(sK[lr * AP + kbk + 1], th, tl); bh[2*v+1] = th; bl[2*v+1] = tl;
    }
    S = __builtin_amdgcn_wmma_f32_16x16x32_bf16(false, ah, false, bh, (short)0, S, false, false);
    S = __builtin_amdgcn_wmma_f32_16x16x32_bf16(false, ah, false, bl, (short)0, S, false, false);
    S = __builtin_amdgcn_wmma_f32_16x16x32_bf16(false, al, false, bh, (short)0, S, false, false);
  }

  // ---- mask + softmax over e (across each 16-lane half)
  const int* mrow = mask + (long)p * 256;
  float pv[8];
  #pragma unroll
  for (int r = 0; r < 8; ++r) {
    int hrow = r + half * 8;
    float s = S[r] * 0.125f;                 // 1/sqrt(64)
    if (mrow[hrow * 16 + lr] == 0) s = -1.0e20f;
    pv[r] = s;
  }
  #pragma unroll
  for (int r = 0; r < 8; ++r) {
    float mx = pv[r];
    mx = fmaxf(mx, __shfl_xor(mx, 1));
    mx = fmaxf(mx, __shfl_xor(mx, 2));
    mx = fmaxf(mx, __shfl_xor(mx, 4));
    mx = fmaxf(mx, __shfl_xor(mx, 8));
    float ex = __expf(pv[r] - mx);
    float sm = ex;
    sm += __shfl_xor(sm, 1);
    sm += __shfl_xor(sm, 2);
    sm += __shfl_xor(sm, 4);
    sm += __shfl_xor(sm, 8);
    pv[r] = ex / sm;
    sP[(r + half * 8) * PP + lr] = pv[r];
  }
  __syncthreads();

  // ---- O = P * v : A = P (16x16 padded to K=32), B = v chunk (16->32 K pad)
  v16bf ph, pl;
  #pragma unroll
  for (int v = 0; v < 8; ++v) {
    if (v < 4) {
      int k0 = half * 8 + 2 * v;     // K in [0,16)
      __bf16 th, tl;
      split2(sP[lr * PP + k0],     th, tl); ph[2*v]   = th; pl[2*v]   = tl;
      split2(sP[lr * PP + k0 + 1], th, tl); ph[2*v+1] = th; pl[2*v+1] = tl;
    } else {                         // K >= 16: zero padding
      ph[2*v] = (__bf16)0.0f; ph[2*v+1] = (__bf16)0.0f;
      pl[2*v] = (__bf16)0.0f; pl[2*v+1] = (__bf16)0.0f;
    }
  }

  float* orow = attout + (long)p * 1024;
  #pragma unroll
  for (int ni = 0; ni < 4; ++ni) {
    v16bf vh, vl;
    #pragma unroll
    for (int v = 0; v < 8; ++v) {
      float x0 = 0.0f, x1 = 0.0f;
      if (v < 4) {                   // v>=4 supply K>=16 -> zeros
        int k0 = half * 8 + 2 * v;
        int d  = ni * 16 + lr;
        x0 = sV[k0 * AP + d];
        x1 = sV[(k0 + 1) * AP + d];
      }
      __bf16 th, tl;
      split2(x0, th, tl); vh[2*v]   = th; vl[2*v]   = tl;
      split2(x1, th, tl); vh[2*v+1] = th; vl[2*v+1] = tl;
    }
    v8f O;
    #pragma unroll
    for (int e = 0; e < 8; ++e) O[e] = 0.0f;
    O = __builtin_amdgcn_wmma_f32_16x16x32_bf16(false, ph, false, vh, (short)0, O, false, false);
    O = __builtin_amdgcn_wmma_f32_16x16x32_bf16(false, ph, false, vl, (short)0, O, false, false);
    O = __builtin_amdgcn_wmma_f32_16x16x32_bf16(false, pl, false, vh, (short)0, O, false, false);
    #pragma unroll
    for (int r = 0; r < 8; ++r)
      orow[(r + half * 8) * 64 + ni * 16 + lr] = O[r];
  }
}

// ---------------------------------------------------------------------------
extern "C" void kernel_launch(void* const* d_in, const int* in_sizes, int n_in,
                              void* d_out, int out_size, void* d_ws, size_t ws_size,
                              hipStream_t stream) {
  const float* values    = (const float*)d_in[0];  // (4, 4096, 1024)
  const float* keys      = (const float*)d_in[1];  // (4, 4096, 1024)
  const float* queries   = (const float*)d_in[2];  // (4, 2048, 1024)
  const int*   mask      = (const int*)  d_in[3];  // (4, 4096, 16, 16)
  const float* condition = (const float*)d_in[4];  // (4, 2048, 512)
  const float* Wc        = (const float*)d_in[5];  // (1024, 512)
  const float* bc        = (const float*)d_in[6];  // (1024,)
  const float* Wo        = (const float*)d_in[7];  // (1024, 1024)
  const float* bo        = (const float*)d_in[8];  // (1024,)
  float* out = (float*)d_out;                      // (4, 4096, 1024)

  float* ws_cond = (float*)d_ws;                    // 8192*1024 f32 (33.5 MB)
  float* ws_att  = ws_cond + (size_t)8192 * 1024;   // 16384*1024 f32 (67 MB)

  // 1) cond_feat = condition @ Wc^T + bc    (M=8192, N=1024, K=512)
  gemm_bt_bias<<<dim3(64, 8), 256, 0, stream>>>(condition, Wc, bc, ws_cond,
                                                8192, 1024, 512);
  // 2) per-position 16x16 attention (one wave per (n, s'))
  attn16<<<dim3(16384), 32, 0, stream>>>(values, keys, queries, mask,
                                         ws_cond, ws_att);
  // 3) out = attout @ Wo^T + bo             (M=16384, N=1024, K=1024)
  gemm_bt_bias<<<dim3(128, 8), 256, 0, stream>>>(ws_att, Wo, bo, out,
                                                 16384, 1024, 1024);
  (void)in_sizes; (void)n_in; (void)out_size; (void)ws_size;
}
